// MultimodalDecoderBlock_16432544874851
// MI455X (gfx1250) — compile-verified, matrix operand order checked by reading, not run on hardware
//
#include <hip/hip_runtime.h>
#include <hip/hip_bf16.h>
#include <math.h>

// ---------------------------------------------------------------------------
// MultimodalDecoderBlock for MI455X (gfx1250, wave32, WMMA).
// All GEMMs run through v_wmma_f32_16x16x32_f16 (f16 in, f32 accumulate).
// Weights pre-packed into WMMA B-fragment order; GEMM waves own a 32x64
// output tile (8 WMMAs per 32-deep k-step, B-fragments amortized over two
// M-subtiles) to minimize L2->VGPR bytes per WMMA.
// ---------------------------------------------------------------------------

typedef _Float16 f16;
typedef __attribute__((ext_vector_type(16))) _Float16 v16h;
typedef __attribute__((ext_vector_type(8)))  _Float16 v8h;
typedef __attribute__((ext_vector_type(8)))  float    v8f;
typedef __attribute__((ext_vector_type(4)))  float    v4f;

#define B_   16
#define S_   512
#define H_   768
#define NH_  12
#define DH_  64
#define I_   3072
#define M_   (B_ * S_)   // 8192 rows in every projection GEMM

static __device__ __forceinline__ v16h cat8(v8h lo, v8h hi) {
  return __builtin_shufflevector(lo, hi, 0,1,2,3,4,5,6,7,8,9,10,11,12,13,14,15);
}

// ---------------- fp32 -> fp16 convert ----------------
__global__ __launch_bounds__(256) void cvt_f32_f16(const float* __restrict__ x,
                                                   f16* __restrict__ y, int n) {
  int i = blockIdx.x * 256 + threadIdx.x;
  if (i < n) y[i] = (f16)x[i];
}

// ---------------- weight packing into WMMA B-fragment order ----------------
// Wp[((kb*(N/16)+nb)*32 + lane)*16 + j] = W[kb*32 + kmap(lane,j), nb*16 + (lane&15)]
// kmap per ISA 16-bit matrix striping: lane<16 -> K {0..7, 16..23},
//                                      lane>=16 -> K {8..15, 24..31}
__global__ __launch_bounds__(256) void pack_weight(const float* __restrict__ W,
                                                   f16* __restrict__ Wp,
                                                   int K, int N) {
  int idx = blockIdx.x * 256 + threadIdx.x;
  if (idx >= K * N) return;
  int j    = idx & 15;
  int l    = (idx >> 4) & 31;
  int frag = idx >> 9;
  int nbs  = N >> 4;
  int nb   = frag % nbs;
  int kb   = frag / nbs;
  int k = kb * 32 + ((j < 8) ? j : j + 8) + ((l < 16) ? 0 : 8);
  int n = nb * 16 + (l & 15);
  Wp[idx] = (f16)W[(size_t)k * N + n];
}

// ---------------- WMMA GEMM: C = A[M,K](f16) x Wp(packed f16) + bias -------
// block = 128 threads (4 waves); wave -> 32 rows x 64 cols:
//   per 32-deep k-step: 2 A-frag loads + 4 B-frag loads -> 8 WMMAs.
// mode: 0 = f32 flat [M,N]      1 = f16 flat [M,N]
//       2 = f16 [B,NH,S,DH]     3 = f16 [B,NH,DH,S]   (S_=512 fixed)
// act : 1 = erf-GELU
__global__ __launch_bounds__(128)
void wmma_gemm(const f16* __restrict__ A, const f16* __restrict__ Wp,
               const float* __restrict__ bias, void* __restrict__ Cout,
               int N, int K, int mode, int act) {
  const int lane = threadIdx.x & 31;
  const int wave = threadIdx.x >> 5;
  const int row0 = blockIdx.x * 128 + wave * 32;
  const int col0 = blockIdx.y * 64;
  const int r    = lane & 15;
  const int koff = (lane < 16) ? 0 : 8;
  const int nbs  = N >> 4;
  const int nb0  = col0 >> 4;

  v8f acc[2][4] = {{v8f{}, v8f{}, v8f{}, v8f{}}, {v8f{}, v8f{}, v8f{}, v8f{}}};

  const f16* arow0 = A + (size_t)(row0 + r) * K + koff;
  const f16* arow1 = arow0 + (size_t)16 * K;
  const int nsteps = K >> 5;
  for (int kb = 0; kb < nsteps; ++kb) {
    const f16* ap0 = arow0 + kb * 32;
    const f16* ap1 = arow1 + kb * 32;
    v16h af0 = cat8(*(const v8h*)(ap0), *(const v8h*)(ap0 + 16));
    v16h af1 = cat8(*(const v8h*)(ap1), *(const v8h*)(ap1 + 16));
    __builtin_prefetch(ap0 + 32, 0, 3);   // WGP-scope prefetch of next k-tile
    __builtin_prefetch(ap1 + 32, 0, 3);

    const f16* bk = Wp + ((size_t)kb * nbs + nb0) * 512 + (size_t)lane * 16;
#pragma unroll
    for (int t = 0; t < 4; ++t) {
      v16h bf = *(const v16h*)(bk + t * 512);
      acc[0][t] = __builtin_amdgcn_wmma_f32_16x16x32_f16(
          false, af0, false, bf, (short)0, acc[0][t], false, false);
      acc[1][t] = __builtin_amdgcn_wmma_f32_16x16x32_f16(
          false, af1, false, bf, (short)0, acc[1][t], false, false);
    }
  }

  const int hi8 = (lane < 16) ? 0 : 8;
  const int cl  = lane & 15;
#pragma unroll
  for (int m = 0; m < 2; ++m) {
#pragma unroll
    for (int t = 0; t < 4; ++t) {
#pragma unroll
      for (int v = 0; v < 8; ++v) {
        int rr = row0 + m * 16 + hi8 + v;
        int cc = col0 + t * 16 + cl;
        float val = acc[m][t][v] + bias[cc];
        if (act) val = 0.5f * val * (1.0f + erff(val * 0.70710678118654752f));
        if (mode == 0) {
          ((float*)Cout)[(size_t)rr * N + cc] = val;
        } else if (mode == 1) {
          ((f16*)Cout)[(size_t)rr * N + cc] = (f16)val;
        } else {
          int b  = rr >> 9, s = rr & 511;        // S_ = 512
          int nh = cc >> 6, d = cc & 63;         // DH_ = 64
          size_t dst = (mode == 2)
              ? (((size_t)b * NH_ + nh) * S_ + s) * DH_ + d
              : (((size_t)b * NH_ + nh) * DH_ + d) * S_ + s;
          ((f16*)Cout)[dst] = (f16)val;
        }
      }
    }
  }
}

// ---------------- attention: one wave handles 16 query rows ----------------
// Qh,Kh: [B,NH,S,DH] f16   Vt: [B,NH,DH,S] f16   mask: [B,1,1,S] f32
// ctx out: [B,S,H] f16.  scores staged in 32KB LDS, shuffle softmax.
__global__ __launch_bounds__(32)
void attn_kernel(const f16* __restrict__ Qh, const f16* __restrict__ Kh,
                 const f16* __restrict__ Vt, const float* __restrict__ mask,
                 f16* __restrict__ ctx) {
  __shared__ float sc[16][512];
  const int bh = blockIdx.x;                  // b*NH + h
  const int q0 = blockIdx.y * 16;
  const int b  = bh / NH_;
  const int h  = bh % NH_;
  const int lane = threadIdx.x;
  const int r    = lane & 15;
  const int koff = (lane < 16) ? 0 : 8;
  const int hi8  = (lane < 16) ? 0 : 8;

  // preload Q fragments for both 32-deep k-steps of DH=64
  const f16* qrow = Qh + ((size_t)bh * S_ + q0 + r) * DH_ + koff;
  v16h qf0 = cat8(*(const v8h*)(qrow),      *(const v8h*)(qrow + 16));
  v16h qf1 = cat8(*(const v8h*)(qrow + 32), *(const v8h*)(qrow + 48));

  const float* mrow = mask + (size_t)b * S_;

  // ---- phase 1: S = Q K^T * 1/8 + mask, into LDS ----
  for (int g = 0; g < 8; ++g) {               // 8 groups x 4 tiles = 512 cols
    v8f acc[4] = {v8f{}, v8f{}, v8f{}, v8f{}};
#pragma unroll
    for (int t = 0; t < 4; ++t) {
      int n = (g * 4 + t) * 16 + (lane & 15); // key index (B-frag column)
      const f16* krow = Kh + ((size_t)bh * S_ + n) * DH_ + koff;
      v16h bf0 = cat8(*(const v8h*)(krow),      *(const v8h*)(krow + 16));
      v16h bf1 = cat8(*(const v8h*)(krow + 32), *(const v8h*)(krow + 48));
      acc[t] = __builtin_amdgcn_wmma_f32_16x16x32_f16(
          false, qf0, false, bf0, (short)0, acc[t], false, false);
      acc[t] = __builtin_amdgcn_wmma_f32_16x16x32_f16(
          false, qf1, false, bf1, (short)0, acc[t], false, false);
    }
#pragma unroll
    for (int t = 0; t < 4; ++t)
#pragma unroll
      for (int v = 0; v < 8; ++v) {
        int rr = v + hi8;
        int cc = (g * 4 + t) * 16 + (lane & 15);
        sc[rr][cc] = acc[t][v] * 0.125f + mrow[cc];
      }
  }
  __syncthreads();

  // ---- phase 2: softmax over 512 per row (wave32 shuffle reduce) ----
  for (int row = 0; row < 16; ++row) {
    float x[16];
    float m = -1e30f;
#pragma unroll
    for (int i = 0; i < 16; ++i) { x[i] = sc[row][i * 32 + lane]; m = fmaxf(m, x[i]); }
#pragma unroll
    for (int off = 16; off > 0; off >>= 1) m = fmaxf(m, __shfl_xor(m, off, 32));
    float s = 0.0f;
#pragma unroll
    for (int i = 0; i < 16; ++i) { x[i] = __expf(x[i] - m); s += x[i]; }
#pragma unroll
    for (int off = 16; off > 0; off >>= 1) s += __shfl_xor(s, off, 32);
    float inv = 1.0f / s;
#pragma unroll
    for (int i = 0; i < 16; ++i) sc[row][i * 32 + lane] = x[i] * inv;
  }
  __syncthreads();

  // ---- phase 3: ctx = P x V ----
  v8f acc[4] = {v8f{}, v8f{}, v8f{}, v8f{}};
  for (int kb = 0; kb < 16; ++kb) {           // K = 512
    const float* prow = &sc[r][kb * 32 + koff];
    v4f p0 = *(const v4f*)(prow);
    v4f p1 = *(const v4f*)(prow + 4);
    v4f p2 = *(const v4f*)(prow + 16);
    v4f p3 = *(const v4f*)(prow + 20);
    v16h af;
#pragma unroll
    for (int j = 0; j < 4; ++j) {
      af[j] = (f16)p0[j]; af[4 + j] = (f16)p1[j];
      af[8 + j] = (f16)p2[j]; af[12 + j] = (f16)p3[j];
    }
#pragma unroll
    for (int t = 0; t < 4; ++t) {
      const f16* vrow = Vt + ((size_t)bh * DH_ + t * 16 + (lane & 15)) * S_
                        + kb * 32 + koff;
      v16h bf = cat8(*(const v8h*)(vrow), *(const v8h*)(vrow + 16));
      acc[t] = __builtin_amdgcn_wmma_f32_16x16x32_f16(
          false, af, false, bf, (short)0, acc[t], false, false);
    }
  }
#pragma unroll
  for (int t = 0; t < 4; ++t)
#pragma unroll
    for (int v = 0; v < 8; ++v) {
      int rr = q0 + v + hi8;
      int cc = h * DH_ + t * 16 + (lane & 15);
      ctx[((size_t)b * S_ + rr) * H_ + cc] = (f16)acc[t][v];
    }
}

// ---------------- fused residual + LayerNorm (H=768 = 3*256) ---------------
__global__ __launch_bounds__(256)
void ln_res(const float* __restrict__ x, const float* __restrict__ res,
            const float* __restrict__ g, const float* __restrict__ bb,
            float* __restrict__ out) {
  __shared__ float red[256];
  const int row = blockIdx.x, tid = threadIdx.x;
  const float* xr = x + (size_t)row * H_;
  const float* rr = res + (size_t)row * H_;
  float v0 = xr[tid] + rr[tid];
  float v1 = xr[tid + 256] + rr[tid + 256];
  float v2 = xr[tid + 512] + rr[tid + 512];
  red[tid] = v0 + v1 + v2;
  __syncthreads();
  for (int k = 128; k > 0; k >>= 1) { if (tid < k) red[tid] += red[tid + k]; __syncthreads(); }
  float mu = red[0] * (1.0f / H_);
  __syncthreads();
  float d0 = v0 - mu, d1 = v1 - mu, d2 = v2 - mu;
  red[tid] = d0 * d0 + d1 * d1 + d2 * d2;
  __syncthreads();
  for (int k = 128; k > 0; k >>= 1) { if (tid < k) red[tid] += red[tid + k]; __syncthreads(); }
  float inv = rsqrtf(red[0] * (1.0f / H_) + 1e-12f);
  float* orow = out + (size_t)row * H_;
  orow[tid]       = d0 * inv * g[tid]       + bb[tid];
  orow[tid + 256] = d1 * inv * g[tid + 256] + bb[tid + 256];
  orow[tid + 512] = d2 * inv * g[tid + 512] + bb[tid + 512];
}

// ---------------------------------------------------------------------------
extern "C" void kernel_launch(void* const* d_in, const int* in_sizes, int n_in,
                              void* d_out, int out_size, void* d_ws, size_t ws_size,
                              hipStream_t stream) {
  (void)in_sizes; (void)n_in; (void)out_size; (void)ws_size;
  const float* enc   = (const float*)d_in[0];
  const float* hid   = (const float*)d_in[1];
  const float* m_sa  = (const float*)d_in[2];   // self-attn uses encoder_attention_mask (repo quirk)
  const float* m_ca  = (const float*)d_in[3];
  const float* sa_qw = (const float*)d_in[4];  const float* sa_qb = (const float*)d_in[5];
  const float* sa_kw = (const float*)d_in[6];  const float* sa_kb = (const float*)d_in[7];
  const float* sa_vw = (const float*)d_in[8];  const float* sa_vb = (const float*)d_in[9];
  const float* sa_ow = (const float*)d_in[10]; const float* sa_ob = (const float*)d_in[11];
  const float* sa_lg = (const float*)d_in[12]; const float* sa_lb = (const float*)d_in[13];
  const float* ca_qw = (const float*)d_in[14]; const float* ca_qb = (const float*)d_in[15];
  const float* ca_kw = (const float*)d_in[16]; const float* ca_kb = (const float*)d_in[17];
  const float* ca_vw = (const float*)d_in[18]; const float* ca_vb = (const float*)d_in[19];
  const float* ti_w  = (const float*)d_in[20]; const float* ti_b  = (const float*)d_in[21];
  const float* out_w = (const float*)d_in[22]; const float* out_b = (const float*)d_in[23];
  const float* ol_g  = (const float*)d_in[24]; const float* ol_b  = (const float*)d_in[25];

  // ---- workspace carve (all sizes multiple of 256B) ----
  char* w = (char*)d_ws;
  size_t off = 0;
  const size_t EH = (size_t)M_ * H_;            // 6,291,456 elements
  #define CARVE(T, name, bytes) T name = (T)(w + off); off += ((size_t)(bytes) + 255) & ~(size_t)255;
  CARVE(f16*,  h16,    EH * 2)
  CARVE(f16*,  e16,    EH * 2)
  CARVE(f16*,  wp_saq, (size_t)H_ * H_ * 2)
  CARVE(f16*,  wp_sak, (size_t)H_ * H_ * 2)
  CARVE(f16*,  wp_sav, (size_t)H_ * H_ * 2)
  CARVE(f16*,  wp_sao, (size_t)H_ * H_ * 2)
  CARVE(f16*,  wp_caq, (size_t)H_ * H_ * 2)
  CARVE(f16*,  wp_cak, (size_t)H_ * H_ * 2)
  CARVE(f16*,  wp_cav, (size_t)H_ * H_ * 2)
  CARVE(f16*,  wp_ti,  (size_t)H_ * I_ * 2)
  CARVE(f16*,  wp_out, (size_t)I_ * H_ * 2)
  CARVE(f16*,  qh,     EH * 2)                  // [B,NH,S,DH]
  CARVE(f16*,  kh,     EH * 2)                  // [B,NH,S,DH]
  CARVE(f16*,  vt,     EH * 2)                  // [B,NH,DH,S]
  CARVE(f16*,  ctx,    EH * 2)                  // [B,S,H]
  CARVE(float*, saout, EH * 4)
  CARVE(f16*,  ti16,   (size_t)M_ * I_ * 2)
  #undef CARVE
  // g1 (f32 GEMM scratch, 25.2MB) aliases qh+kh — disjoint lifetimes:
  // qh/kh live only between projection-GEMM and attention of each phase.
  float* g1 = (float*)qh;

  dim3 b256(256);
  // 1) activations -> f16
  cvt_f32_f16<<<(EH + 255) / 256, b256, 0, stream>>>(hid, h16, (int)EH);
  cvt_f32_f16<<<(EH + 255) / 256, b256, 0, stream>>>(enc, e16, (int)EH);
  // 2) pack all weights into WMMA fragment order
  const int gHH = (H_ * H_) / 256, gHI = (H_ * I_) / 256;
  pack_weight<<<gHH, b256, 0, stream>>>(sa_qw, wp_saq, H_, H_);
  pack_weight<<<gHH, b256, 0, stream>>>(sa_kw, wp_sak, H_, H_);
  pack_weight<<<gHH, b256, 0, stream>>>(sa_vw, wp_sav, H_, H_);
  pack_weight<<<gHH, b256, 0, stream>>>(sa_ow, wp_sao, H_, H_);
  pack_weight<<<gHH, b256, 0, stream>>>(ca_qw, wp_caq, H_, H_);
  pack_weight<<<gHH, b256, 0, stream>>>(ca_kw, wp_cak, H_, H_);
  pack_weight<<<gHH, b256, 0, stream>>>(ca_vw, wp_cav, H_, H_);
  pack_weight<<<gHI, b256, 0, stream>>>(ti_w,  wp_ti,  H_, I_);
  pack_weight<<<gHI, b256, 0, stream>>>(out_w, wp_out, I_, H_);

  dim3 gblk(128);
  dim3 ggH(M_ / 128, H_ / 64), ggI(M_ / 128, I_ / 64);
  dim3 ga(B_ * NH_, S_ / 16);

  // 3) self-attention
  wmma_gemm<<<ggH, gblk, 0, stream>>>(h16, wp_saq, sa_qb, qh, H_, H_, 2, 0);
  wmma_gemm<<<ggH, gblk, 0, stream>>>(h16, wp_sak, sa_kb, kh, H_, H_, 2, 0);
  wmma_gemm<<<ggH, gblk, 0, stream>>>(h16, wp_sav, sa_vb, vt, H_, H_, 3, 0);
  attn_kernel<<<ga, 32, 0, stream>>>(qh, kh, vt, m_sa, ctx);
  wmma_gemm<<<ggH, gblk, 0, stream>>>(ctx, wp_sao, sa_ob, g1, H_, H_, 0, 0);
  ln_res<<<M_, b256, 0, stream>>>(g1, hid, sa_lg, sa_lb, saout);

  // 4) cross-attention (Q from decoder, K/V from encoder)
  wmma_gemm<<<ggH, gblk, 0, stream>>>(h16, wp_caq, ca_qb, qh, H_, H_, 2, 0);
  wmma_gemm<<<ggH, gblk, 0, stream>>>(e16, wp_cak, ca_kb, kh, H_, H_, 2, 0);
  wmma_gemm<<<ggH, gblk, 0, stream>>>(e16, wp_cav, ca_vb, vt, H_, H_, 3, 0);
  attn_kernel<<<ga, 32, 0, stream>>>(qh, kh, vt, m_ca, ctx);

  // 5) FFN: erf-GELU fused into first GEMM, f16 output feeds second GEMM
  wmma_gemm<<<ggI, gblk, 0, stream>>>(ctx,  wp_ti,  ti_b,  ti16, I_, H_, 1, 1);
  wmma_gemm<<<ggH, gblk, 0, stream>>>(ti16, wp_out, out_b, g1,   H_, I_, 0, 0);
  ln_res<<<M_, b256, 0, stream>>>(g1, saout, ol_g, ol_b, (float*)d_out);
}
